// KronLinear_45921790329290
// MI455X (gfx1250) — matrix operand order repaired and checked by reading, not compiled, not run on hardware
//
#include <hip/hip_runtime.h>

// ---------------------------------------------------------------------------
// KronLinear forward for MI455X (gfx1250, wave32, WMMA bf16 16x16x32)
//
// out[n, a2*64+b2] = bias + sum_r sum_a1 A[r,a1,a2] * T[n,a1,r*64+b2]
// T[n,a1,c]        = sum_b1 x[n,a1*64+b1] * Beff[b1,c]
// Beff[b1,r*64+b2] = B[b2%4, b1, r*16 + b2/4]   (faithful to the torch .view)
// ---------------------------------------------------------------------------

typedef __attribute__((ext_vector_type(16))) __bf16 v16bf;
typedef __attribute__((ext_vector_type(8)))  float  v8f;

#define DIM   4096
#define NROWS 8192           // 4 * 2048
#define RPB   16             // rows per block
#define LROW  72             // padded LDS row stride (bf16 elems); 36 dwords -> bank friendly

// LDS layout (bf16 rows of LROW unless noted):
//   sBeff : 256 rows   (BeffT[c][b1])
//   sAt   : 256 rows   (At[r*64+a2][a1])
//   sX    :  64 rows   (x row, bf16, [a1][b1])
//   sT    : 256 rows   (T transposed: [c][a1], bf16)
//   sBias : 4096 f32
#define SMEM_BYTES ((256 + 256 + 64 + 256) * LROW * 2 + 4096 * 4)

// ---------------------------- prep kernel ----------------------------------
__global__ void kron_prep(const float* __restrict__ A, const float* __restrict__ B,
                          __bf16* __restrict__ BeffT, __bf16* __restrict__ At) {
    int i = blockIdx.x * blockDim.x + threadIdx.x;     // 0 .. 16383
    if (i >= 256 * 64) return;
    // BeffT[c][b1], c = r*64 + b2
    {
        int c  = i >> 6;
        int b1 = i & 63;
        int r  = c >> 6;
        int b2 = c & 63;
        BeffT[i] = (__bf16)B[((b2 & 3) * 64 + b1) * 64 + (r * 16 + (b2 >> 2))];
    }
    // At[r*64+a2][a1] = A[r][a1][a2]
    {
        int row = i >> 6;                  // r*64 + a2
        int a1  = i & 63;
        int r   = row >> 6;
        int a2  = row & 63;
        At[i] = (__bf16)A[(r * 64 + a1) * 64 + a2];
    }
}

// operand fetch helpers (16-B LDS loads matching the documented WMMA layouts)
__device__ __forceinline__ v16bf ld_aop(const __bf16* row, int kb, int hi) {
    // A-operand: lanes<16 hold K{kb+0..7, kb+16..23}; lanes>=16 K{kb+8..15, kb+24..31}
    v16bf v;
    const int oa = kb + hi * 8;
    ((uint4*)&v)[0] = *(const uint4*)(row + oa);
    ((uint4*)&v)[1] = *(const uint4*)(row + oa + 16);
    return v;
}
__device__ __forceinline__ v16bf ld_bop(const __bf16* row, int kb, int hi) {
    // B-operand: lanes<16 hold K{kb+0..7, kb+16..23}; lanes>=16 K{kb+8..15, kb+24..31}
    v16bf v;
    const int ob = kb + hi * 8;
    ((uint4*)&v)[0] = *(const uint4*)(row + ob);
    ((uint4*)&v)[1] = *(const uint4*)(row + ob + 16);
    return v;
}

#define WMMA_BF16(a, b, c) \
    __builtin_amdgcn_wmma_f32_16x16x32_bf16(false, (a), false, (b), (short)0, (c), false, false)

// ---------------------------- main kernel ----------------------------------
__launch_bounds__(256, 2)
__global__ void kron_main(const float* __restrict__ x, const float* __restrict__ bias,
                          const __bf16* __restrict__ BeffT, const __bf16* __restrict__ At,
                          float* __restrict__ out) {
    extern __shared__ char smem_raw[];
    __bf16* sBeff = (__bf16*)smem_raw;
    __bf16* sAt   = sBeff + 256 * LROW;
    __bf16* sX    = sAt   + 256 * LROW;
    __bf16* sT    = sX    +  64 * LROW;
    float*  sBias = (float*)(sT + 256 * LROW);

    const int tid  = threadIdx.x;
    const int lane = tid & 31;
    const int wave = tid >> 5;
    const int lh   = lane & 15;     // column / row index within 16x16 tile
    const int hi   = lane >> 4;     // lane-half select

    // ---- one-time staging of weights (bf16) and bias (f32) into LDS ----
    for (int r = tid; r < 512; r += 256) {
        const __bf16* src = (r < 256) ? (BeffT + r * 64) : (At + (r - 256) * 64);
        __bf16*       dst = (r < 256) ? (sBeff + r * LROW) : (sAt + (r - 256) * LROW);
#pragma unroll
        for (int j = 0; j < 8; ++j)
            *(uint4*)(dst + j * 8) = *(const uint4*)(src + j * 8);
    }
    {
        const float4* bsrc = (const float4*)bias;
        float4*       bdst = (float4*)sBias;
#pragma unroll
        for (int j = 0; j < 4; ++j)
            bdst[tid + 256 * j] = bsrc[tid + 256 * j];
    }
    __syncthreads();

    const int n0 = blockIdx.x * RPB;

    for (int ni = 0; ni < RPB; ++ni) {
        const int n = n0 + ni;

        // ---------------- Phase A: stage x[n] (f32) -> sX (bf16) ----------
        {
            const float4* xr4 = (const float4*)(x + (size_t)n * DIM + tid * 16);
            if (ni + 1 < RPB)   // pull next row toward L2/WGP$ while we compute
                __builtin_prefetch(x + (size_t)(n + 1) * DIM + tid * 16, 0, 0);
            alignas(16) __bf16 h[16];
#pragma unroll
            for (int j = 0; j < 4; ++j) {
                float4 f = xr4[j];
                h[j * 4 + 0] = (__bf16)f.x;
                h[j * 4 + 1] = (__bf16)f.y;
                h[j * 4 + 2] = (__bf16)f.z;
                h[j * 4 + 3] = (__bf16)f.w;
            }
            __bf16* dst = sX + (tid >> 2) * LROW + (tid & 3) * 16;
            *(uint4*)(dst + 0) = *(uint4*)&h[0];
            *(uint4*)(dst + 8) = *(uint4*)&h[8];
        }
        __syncthreads();   // sX ready; also protects sX/sT vs previous row

        // ---------------- Phase B: T(64x256) = X(64x64) @ Beff(64x256) ----
        // Two tiles per step sharing the A-operand (same a1 tile), separate
        // accumulators -> independent WMMAs fill each other's hazard slots.
#pragma unroll
        for (int mt = 0; mt < 4; ++mt) {
            const int ct0 = wave;           // column tiles w and w+8
            const int ct1 = wave + 8;
            const __bf16* xrow  = sX    + (mt  * 16 + lh) * LROW;
            const __bf16* brow0 = sBeff + (ct0 * 16 + lh) * LROW;
            const __bf16* brow1 = sBeff + (ct1 * 16 + lh) * LROW;
            v8f acc0 = {0.f, 0.f, 0.f, 0.f, 0.f, 0.f, 0.f, 0.f};
            v8f acc1 = {0.f, 0.f, 0.f, 0.f, 0.f, 0.f, 0.f, 0.f};
#pragma unroll
            for (int kt = 0; kt < 2; ++kt) {
                const int kb = kt * 32;
                const v16bf av  = ld_aop(xrow,  kb, hi);
                const v16bf bv0 = ld_bop(brow0, kb, hi);
                const v16bf bv1 = ld_bop(brow1, kb, hi);
                acc0 = WMMA_BF16(av, bv0, acc0);
                acc1 = WMMA_BF16(av, bv1, acc1);
            }
            // C/D layout: elem j is (m = j + 8*hi, n = lh). Transpose-store.
            alignas(16) __bf16 t0[8], t1[8];
#pragma unroll
            for (int j = 0; j < 8; ++j) { t0[j] = (__bf16)acc0[j]; t1[j] = (__bf16)acc1[j]; }
            *(uint4*)(sT + (ct0 * 16 + lh) * LROW + mt * 16 + hi * 8) = *(uint4*)t0;
            *(uint4*)(sT + (ct1 * 16 + lh) * LROW + mt * 16 + hi * 8) = *(uint4*)t1;
        }
        __syncthreads();   // sT ready for phase C; sX free for next row

        // ---------------- Phase C: Out(64x64) = sum_r At_r @ U_r ----------
        // Each wave owns tiles (2w, 2w+1): same a2 tile (shared A-operand),
        // adjacent b2 tiles; rank accumulates in the f32 accumulators.
        {
            const int mt  = wave >> 1;          // a2 tile 0..3
            const int nt0 = (2 * wave) & 3;     // b2 tiles
            const int nt1 = nt0 + 1;
            v8f acc0 = {0.f, 0.f, 0.f, 0.f, 0.f, 0.f, 0.f, 0.f};
            v8f acc1 = {0.f, 0.f, 0.f, 0.f, 0.f, 0.f, 0.f, 0.f};
#pragma unroll
            for (int r = 0; r < 4; ++r) {
                const __bf16* arow  = sAt + (r * 64 + mt  * 16 + lh) * LROW;
                const __bf16* brow0 = sT  + (r * 64 + nt0 * 16 + lh) * LROW;
                const __bf16* brow1 = sT  + (r * 64 + nt1 * 16 + lh) * LROW;
#pragma unroll
                for (int kt = 0; kt < 2; ++kt) {
                    const int kb = kt * 32;
                    const v16bf av  = ld_aop(arow,  kb, hi);
                    const v16bf bv0 = ld_bop(brow0, kb, hi);
                    const v16bf bv1 = ld_bop(brow1, kb, hi);
                    acc0 = WMMA_BF16(av, bv0, acc0);
                    acc1 = WMMA_BF16(av, bv1, acc1);
                }
            }
            // epilogue: out[n, a2*64+b2] = acc + bias
            const int b2_0 = nt0 * 16 + lh;
            const int b2_1 = nt1 * 16 + lh;
            float* orow = out + (size_t)n * DIM;
#pragma unroll
            for (int j = 0; j < 8; ++j) {
                const int a2   = mt * 16 + hi * 8 + j;
                const int idx0 = a2 * 64 + b2_0;
                const int idx1 = a2 * 64 + b2_1;
                orow[idx0] = acc0[j] + sBias[idx0];
                orow[idx1] = acc1[j] + sBias[idx1];
            }
        }
        // no barrier needed here: next row's post-staging barrier separates
        // this phase's sT reads from the next phase-B sT writes.
    }
}

// ---------------------------- launcher -------------------------------------
extern "C" void kernel_launch(void* const* d_in, const int* in_sizes, int n_in,
                              void* d_out, int out_size, void* d_ws, size_t ws_size,
                              hipStream_t stream) {
    const float* x    = (const float*)d_in[0];
    const float* A    = (const float*)d_in[1];
    const float* B    = (const float*)d_in[2];
    const float* bias = (const float*)d_in[3];
    float*       out  = (float*)d_out;

    __bf16* BeffT = (__bf16*)d_ws;          // 256*64 bf16
    __bf16* At    = BeffT + 256 * 64;       // 256*64 bf16   (64 KB total)

    kron_prep<<<64, 256, 0, stream>>>(A, B, BeffT, At);
    kron_main<<<NROWS / RPB, 256, SMEM_BYTES, stream>>>(x, bias, BeffT, At, out);
}
